// MultiHeadAttention_19816979103858
// MI455X (gfx1250) — compile-verified, matrix operand order checked by reading, not run on hardware
//
#include <hip/hip_runtime.h>

#define B_  4
#define S_  2048
#define D_  1024
#define H_  16
#define DK_ 64

typedef __bf16 bf16_t;
typedef __attribute__((ext_vector_type(16))) __bf16 v16bf;
typedef __attribute__((ext_vector_type(8)))  __bf16 v8bf;
typedef __attribute__((ext_vector_type(8)))  float  v8f;
typedef __attribute__((ext_vector_type(4)))  unsigned int u32x4;
typedef __attribute__((ext_vector_type(8)))  int i32x8;
typedef __attribute__((ext_vector_type(4)))  int i32x4;

#if __has_builtin(__builtin_amdgcn_tensor_load_to_lds) && __has_builtin(__builtin_amdgcn_s_wait_tensorcnt)
#define HAVE_TDM 1
#else
#define HAVE_TDM 0
#endif

static __device__ __forceinline__ v8f wmma_bf16(v16bf a, v16bf b, v8f c) {
  // D = A(16x32 bf16) * B(32x16 bf16) + C(16x16 f32)
  return __builtin_amdgcn_wmma_f32_16x16x32_bf16(false, a, false, b, (short)0, c, false, false);
}

// Fragment = two contiguous 8-element bf16 runs: [p, p+7] and [p+16, p+23].
// Matches 16-bit A/B VGPR layout with kb0 already folded into p.
static __device__ __forceinline__ v16bf frag2x8(const bf16_t* p) {
  v8bf a = *(const v8bf*)p;
  v8bf b = *(const v8bf*)(p + 16);
  v16bf r;
#pragma unroll
  for (int e = 0; e < 8; ++e) { r[e] = a[e]; r[e + 8] = b[e]; }
  return r;
}

// ---------------------------------------------------------------------------
// GEMM: out[M x 1024] = A[M x 1024] (fp32) @ W[1024 x 1024] (fp32) + bias
// mode 0: write bf16, permuted to [B, H, S, DK]   (projections)
// mode 1: write fp32, row-major [M x 1024]        (output projection)
// Block: 256 threads = 8 waves; wave tile 16x64; block tile 128x64.
// k-step 64, software-pipelined W-tile staging; B-fragments preloaded so the
// 4 WMMAs per k-chunk issue back-to-back.
// ---------------------------------------------------------------------------
#define WT_STRIDE 72   // padded [n][k] row stride (multiple of 8, bank-friendly)

__global__ __launch_bounds__(256)
void gemm_kernel(const float* __restrict__ A,
                 const float* __restrict__ W,
                 const float* __restrict__ bias,
                 bf16_t* __restrict__ outPerm,
                 float*  __restrict__ outPlain,
                 int mode)
{
  __shared__ bf16_t Wt[64 * WT_STRIDE];    // transposed W tile [n_local][k_local]

  const int tid    = threadIdx.x;
  const int lane   = tid & 31;
  const int wave   = tid >> 5;
  const int l16    = lane & 15;
  const int halfHi = lane >> 4;            // 0: lanes 0-15, 1: lanes 16-31
  const int kb0    = halfHi ? 8 : 0;

  const int n0 = blockIdx.x * 64;
  const int m0 = blockIdx.y * 128 + wave * 16;

  // staging mapping: thread -> (k row, 16-column chunk)
  const int sk    = tid >> 2;              // 0..63
  const int spart = tid & 3;               // 0..3 -> 16 cols each

  v8f acc[4];
#pragma unroll
  for (int t = 0; t < 4; ++t)
#pragma unroll
    for (int i = 0; i < 8; ++i) acc[t][i] = 0.0f;

  const float* Arow = A + (size_t)(m0 + l16) * D_;

  float regs[16];
  {
    const float* wp = W + (size_t)sk * D_ + n0 + spart * 16;
#pragma unroll
    for (int j = 0; j < 16; ++j) regs[j] = wp[j];
  }

  for (int it = 0; it < D_ / 64; ++it) {
    __syncthreads();                       // previous tile's readers done
    // ---- store staged regs transposed into LDS (fp32 -> bf16) ----
#pragma unroll
    for (int j = 0; j < 16; ++j)
      Wt[(spart * 16 + j) * WT_STRIDE + sk] = (bf16_t)regs[j];
    __syncthreads();

    // ---- issue global loads for next tile; they overlap the WMMAs below ----
    if (it + 1 < D_ / 64) {
      const float* wp = W + (size_t)((it + 1) * 64 + sk) * D_ + n0 + spart * 16;
#pragma unroll
      for (int j = 0; j < 16; ++j) regs[j] = wp[j];
    }

    const int k0 = it * 64;
#pragma unroll
    for (int kc = 0; kc < 2; ++kc) {
      // A fragment (fp32 -> bf16 in-register)
      v16bf afrag;
      {
        const float* ap = Arow + k0 + kc * 32 + kb0;
#pragma unroll
        for (int e = 0; e < 8; ++e) { afrag[e] = (bf16_t)ap[e]; afrag[e + 8] = (bf16_t)ap[16 + e]; }
      }
      // preload all B fragments, then issue WMMAs back-to-back
      v16bf bfr[4];
#pragma unroll
      for (int t = 0; t < 4; ++t)
        bfr[t] = frag2x8(&Wt[(t * 16 + l16) * WT_STRIDE + kc * 32 + kb0]);
#pragma unroll
      for (int t = 0; t < 4; ++t)
        acc[t] = wmma_bf16(afrag, bfr[t], acc[t]);
    }
  }

  // ---- epilogue: bias + store ----
#pragma unroll
  for (int t = 0; t < 4; ++t) {
    const int c  = n0 + t * 16 + l16;
    const float bv = bias[c];
#pragma unroll
    for (int i = 0; i < 8; ++i) {
      const int r = m0 + i + halfHi * 8;
      const float val = acc[t][i] + bv;
      if (mode == 0) {
        const int b = r >> 11, s = r & (S_ - 1);
        const int h = c >> 6, dk = c & (DK_ - 1);
        outPerm[(((size_t)b * H_ + h) * S_ + s) * DK_ + dk] = (bf16_t)val;
      } else {
        outPlain[(size_t)r * D_ + c] = val;
      }
    }
  }
}

// ---------------------------------------------------------------------------
// Flash attention, causal. Qp/Kp/Vp are bf16 [B*H, S, DK].
// Block: 256 threads = 8 waves; wave = 16 queries; key blocks of 32.
// K tile staged via TDM (tensor_load_to_lds); V staged transposed.
// attn_v fp32 [B, S, H*DK].
// ---------------------------------------------------------------------------
#define VT_STRIDE 40   // padded [dk][key] row stride
#define P_STRIDE  40   // padded per-wave P row stride

__global__ __launch_bounds__(256)
void attn_kernel(const bf16_t* __restrict__ Qp,
                 const bf16_t* __restrict__ Kp,
                 const bf16_t* __restrict__ Vp,
                 float* __restrict__ attnv)
{
  __shared__ bf16_t Klds[32 * 64];                 // [key][dk]
  __shared__ bf16_t Vt[64 * VT_STRIDE];            // [dk][key] transposed
  __shared__ bf16_t Plds[8][16 * P_STRIDE];        // per-wave P tile [row][key]

  const int tid    = threadIdx.x;
  const int lane   = tid & 31;
  const int wave   = tid >> 5;
  const int l16    = lane & 15;
  const int halfHi = lane >> 4;
  const int kb0    = halfHi ? 8 : 0;

  const int h  = blockIdx.y;
  const int b  = blockIdx.z;
  const int bh = b * H_ + h;
  const int qblk = blockIdx.x;             // 128 queries per block
  const int q0   = qblk * 128 + wave * 16; // wave's query base

  // ---- preload Q fragments (dk chunks 0..31 and 32..63) ----
  const bf16_t* Qbase = Qp + ((size_t)bh * S_ + q0 + l16) * DK_;
  v16bf qf[2];
#pragma unroll
  for (int kc = 0; kc < 2; ++kc) qf[kc] = frag2x8(Qbase + kc * 32 + kb0);

  v8f acc[4];
  float mrow[8], lrow[8];
#pragma unroll
  for (int t = 0; t < 4; ++t)
#pragma unroll
    for (int i = 0; i < 8; ++i) acc[t][i] = 0.0f;
#pragma unroll
  for (int i = 0; i < 8; ++i) { mrow[i] = -1e30f; lrow[i] = 0.0f; }

  const float scale = 0.125f;              // 1/sqrt(64)
  const int nkb = (qblk + 1) * 4;          // uniform across waves in block

  for (int kb = 0; kb < nkb; ++kb) {
    const int key0 = kb * 32;
    __syncthreads();                       // protect LDS from previous iteration

#if HAVE_TDM
    // ---- K tile: contiguous 4KB copy via Tensor Data Mover ----
    if (wave == 0) {
      const unsigned long long ga =
          (unsigned long long)(uintptr_t)(Kp + ((size_t)bh * S_ + key0) * DK_);
      const unsigned int lds = (unsigned int)(uintptr_t)&Klds[0];
      u32x4 g0;
      g0[0] = 1u;                                          // count=1
      g0[1] = lds;                                         // lds_addr
      g0[2] = (unsigned int)ga;                            // global_addr[31:0]
      g0[3] = ((unsigned int)(ga >> 32) & 0x01FFFFFFu) | (2u << 30); // addr[56:32] | type=2
      i32x8 g1;
      g1[0] = (int)(1u << 16);                             // data_size=1 (2 bytes)
      g1[1] = (int)(2048u << 16);                          // tensor_dim0 = 2048 (low 16)
      g1[2] = 0;                                           // tensor_dim0 hi / tensor_dim1 lo
      g1[3] = (int)(2048u << 16);                          // tile_dim0 = 2048
      g1[4] = 0; g1[5] = 0; g1[6] = 0; g1[7] = 0;
      i32x4 gz4; gz4[0] = 0; gz4[1] = 0; gz4[2] = 0; gz4[3] = 0;
      i32x8 gz8;
#pragma unroll
      for (int j = 0; j < 8; ++j) gz8[j] = 0;
      __builtin_amdgcn_tensor_load_to_lds(g0, g1, gz4, gz4, gz8, 0);
    }
#endif
    // ---- V tile transposed: vector global loads + scalar LDS stores ----
    {
      const int key  = tid >> 3;           // 0..31
      const int part = tid & 7;            // 0..7
      const bf16_t* vp = Vp + ((size_t)bh * S_ + key0 + key) * DK_ + part * 8;
      v8bf vv = *(const v8bf*)vp;
#pragma unroll
      for (int j = 0; j < 8; ++j) Vt[(part * 8 + j) * VT_STRIDE + key] = vv[j];
#if !HAVE_TDM
      const bf16_t* kp = Kp + ((size_t)bh * S_ + key0 + key) * DK_ + part * 8;
      *(v8bf*)&Klds[key * 64 + part * 8] = *(const v8bf*)kp;
#endif
    }
#if HAVE_TDM
    if (wave == 0) __builtin_amdgcn_s_wait_tensorcnt(0);
    asm volatile("" ::: "memory");
#endif
    __syncthreads();

    // ---- scores: preload 4 K fragments, then 4 WMMAs (2 indep. chains) ----
    v8f s0, s1;
#pragma unroll
    for (int i = 0; i < 8; ++i) { s0[i] = 0.0f; s1[i] = 0.0f; }
    {
      v16bf kf0a = frag2x8(&Klds[l16 * 64 + kb0]);
      v16bf kf0b = frag2x8(&Klds[(16 + l16) * 64 + kb0]);
      v16bf kf1a = frag2x8(&Klds[l16 * 64 + 32 + kb0]);
      v16bf kf1b = frag2x8(&Klds[(16 + l16) * 64 + 32 + kb0]);
      s0 = wmma_bf16(qf[0], kf0a, s0);
      s1 = wmma_bf16(qf[0], kf0b, s1);
      s0 = wmma_bf16(qf[1], kf1a, s0);
      s1 = wmma_bf16(qf[1], kf1b, s1);
    }

    // ---- scale + causal mask + online softmax ----
    float p0[8], p1[8];
#pragma unroll
    for (int i = 0; i < 8; ++i) {
      const int q = q0 + i + halfHi * 8;
      p0[i] = (key0 + l16      <= q) ? s0[i] * scale : -1e30f;
      p1[i] = (key0 + 16 + l16 <= q) ? s1[i] * scale : -1e30f;
    }
#pragma unroll
    for (int i = 0; i < 8; ++i) {
      float rmax = fmaxf(p0[i], p1[i]);
#pragma unroll
      for (int off = 1; off < 16; off <<= 1)
        rmax = fmaxf(rmax, __shfl_xor(rmax, off, 32));
      const float mnew  = fmaxf(mrow[i], rmax);
      const float alpha = __expf(mrow[i] - mnew);
      mrow[i] = mnew;
      p0[i] = __expf(p0[i] - mnew);
      p1[i] = __expf(p1[i] - mnew);
      float rs = p0[i] + p1[i];
#pragma unroll
      for (int off = 1; off < 16; off <<= 1)
        rs += __shfl_xor(rs, off, 32);
      lrow[i] = lrow[i] * alpha + rs;
#pragma unroll
      for (int t = 0; t < 4; ++t) acc[t][i] *= alpha;
    }

    // ---- re-swizzle P (C layout) -> A layout via per-wave LDS tile ----
    bf16_t* Pw = Plds[wave];
#pragma unroll
    for (int i = 0; i < 8; ++i) {
      const int M = i + halfHi * 8;
      Pw[M * P_STRIDE + l16]      = (bf16_t)p0[i];
      Pw[M * P_STRIDE + 16 + l16] = (bf16_t)p1[i];
    }
    __syncthreads();

    // ---- P·V: preload all fragments, issue 4 WMMAs back-to-back ----
    {
      v16bf pf = frag2x8(&Pw[l16 * P_STRIDE + kb0]);
      v16bf vfr[4];
#pragma unroll
      for (int t = 0; t < 4; ++t)
        vfr[t] = frag2x8(&Vt[(t * 16 + l16) * VT_STRIDE + kb0]);
#pragma unroll
      for (int t = 0; t < 4; ++t)
        acc[t] = wmma_bf16(pf, vfr[t], acc[t]);
    }
  }

  // ---- finalize: divide by row sum, write attn_v [b, q, h*64 + dk] fp32 ----
#pragma unroll
  for (int t = 0; t < 4; ++t) {
#pragma unroll
    for (int i = 0; i < 8; ++i) {
      const int q  = q0 + i + halfHi * 8;
      const int dk = t * 16 + l16;
      attnv[((size_t)b * S_ + q) * D_ + h * DK_ + dk] = acc[t][i] / lrow[i];
    }
  }
}

// ---------------------------------------------------------------------------
extern "C" void kernel_launch(void* const* d_in, const int* in_sizes, int n_in,
                              void* d_out, int out_size, void* d_ws, size_t ws_size,
                              hipStream_t stream) {
  (void)in_sizes; (void)n_in; (void)out_size; (void)ws_size;
  const float* query = (const float*)d_in[0];
  const float* key   = (const float*)d_in[1];
  const float* value = (const float*)d_in[2];
  // d_in[3] = causal bool mask: recomputed analytically in-kernel
  const float* Wq = (const float*)d_in[4];
  const float* bq = (const float*)d_in[5];
  const float* Wk = (const float*)d_in[6];
  const float* bk = (const float*)d_in[7];
  const float* Wv = (const float*)d_in[8];
  const float* bv = (const float*)d_in[9];
  const float* Wc = (const float*)d_in[10];
  const float* bc = (const float*)d_in[11];

  float* out   = (float*)d_out;                         // output  [B,S,D]
  float* attnv = out + (size_t)B_ * S_ * D_;            // attn_v  [B,S,D]

  const size_t nProj = (size_t)B_ * H_ * S_ * DK_;      // 8,388,608 elems
  bf16_t* Qp = (bf16_t*)d_ws;
  bf16_t* Kp = Qp + nProj;
  bf16_t* Vp = Kp + nProj;

  const dim3 gGemm(D_ / 64, (B_ * S_) / 128);           // (16, 64)
  const dim3 blk(256);

  gemm_kernel<<<gGemm, blk, 0, stream>>>(query, Wq, bq, Qp, nullptr, 0);
  gemm_kernel<<<gGemm, blk, 0, stream>>>(key,   Wk, bk, Kp, nullptr, 0);
  gemm_kernel<<<gGemm, blk, 0, stream>>>(value, Wv, bv, Vp, nullptr, 0);

  attn_kernel<<<dim3(S_ / 128, H_, B_), blk, 0, stream>>>(Qp, Kp, Vp, attnv);

  gemm_kernel<<<gGemm, blk, 0, stream>>>(attnv, Wc, bc, nullptr, out, 1);
}
